// LS2Actor_79001628443221
// MI455X (gfx1250) — matrix operand order repaired and checked by reading, not compiled, not run on hardware
//
#include <hip/hip_runtime.h>

typedef __bf16 bf16_t;
typedef bf16_t bf16x4  __attribute__((ext_vector_type(4)));
typedef bf16_t bf16x8  __attribute__((ext_vector_type(8)));
typedef bf16_t bf16x16 __attribute__((ext_vector_type(16)));
typedef float  v8f     __attribute__((ext_vector_type(8)));

#define DM  128
#define K3  384
#define GN_EPS 1e-5f

// ---------------------------------------------------------------------------
// WMMA helpers (CDNA5 wave32, v_wmma_f32_16x16x32_bf16)
// ---------------------------------------------------------------------------
__device__ __forceinline__ v8f wmma_bf16(bf16x16 a, bf16x16 b, v8f c) {
  return __builtin_amdgcn_wmma_f32_16x16x32_bf16(
      /*neg_a=*/false, a, /*neg_b=*/false, b,
      /*c_mod=*/(short)0, c, /*reuse_a=*/false, /*reuse_b=*/false);
}

// Async memory->LDS copy (16B per lane), tracked by ASYNCcnt.
// lds_off is the byte offset from the workgroup LDS base (dynamic LDS = 0).
__device__ __forceinline__ void async_gather_b128(unsigned lds_off, const void* gaddr) {
  asm volatile("global_load_async_to_lds_b128 %0, %1, off"
               :: "v"(lds_off), "v"(gaddr)
               : "memory");
}
__device__ __forceinline__ void wait_async0() {
  asm volatile("s_wait_asynccnt 0x0" ::: "memory");
}

// A fragment (16x32 bf16, MxK), row-major source with leading dim `ld`.
// ISA layout: lanes 0-15 row M=lane, K={0..7,16..23}; lanes 16-31 K={8..15,24..31}.
__device__ __forceinline__ bf16x16 load_a_frag(const bf16_t* base, int ld, int k0, int lane) {
  const int r  = lane & 15;
  const int hk = (lane >> 4) << 3;
  const bf16_t* p = base + (long)r * ld + k0 + hk;
  bf16x8 lo = *(const bf16x8*)p;
  bf16x8 hi = *(const bf16x8*)(p + 16);
  bf16x16 o;
#pragma unroll
  for (int j = 0; j < 8; ++j) { o[j] = lo[j]; o[j + 8] = hi[j]; }
  return o;
}

// B fragment (32x16 bf16, KxN) for X @ W.T: B[k][n] = W[n][k], W row-major [Dout,K].
// Lanes 0-15: N=lane, K=k0+{0..7,16..23} ; lanes 16-31: N=lane-16, K=k0+{8..15,24..31}.
__device__ __forceinline__ bf16x16 load_b_frag(const bf16_t* W, int K, int nt, int k0, int lane) {
  const int n = (nt << 4) + (lane & 15);
  const bf16_t* p = W + (long)n * K + k0 + ((lane >> 4) << 3);
  bf16x8 lo = *(const bf16x8*)p;
  bf16x8 hi = *(const bf16x8*)(p + 16);
  bf16x16 o;
#pragma unroll
  for (int j = 0; j < 8; ++j) { o[j] = lo[j]; o[j + 8] = hi[j]; }
  return o;
}

// Stage 8 accumulator fragments (a full 16x128 f32 tile) into LDS.
// C/D layout: VGPR v, lane l -> row m = v + 8*(l>>4), col n = ntile*16 + (l&15).
__device__ __forceinline__ void stage_acc(float* tile, const v8f* acc, int lane) {
  const int m0 = (lane >> 4) << 3;
  const int n0 = lane & 15;
#pragma unroll
  for (int nt = 0; nt < 8; ++nt)
#pragma unroll
    for (int v = 0; v < 8; ++v)
      tile[(m0 + v) * DM + (nt << 4) + n0] = acc[nt][v];
}

// GroupNorm(1 group over 128 ch) on a 16x128 f32 LDS tile; writes bf16 rows to
// dst (LDS or global), leading dim ldd. Lane l handles row (l&15), half (l>>4);
// halves combined with a cross-half shuffle.
__device__ __forceinline__ void gn_from_tile(const float* tile, const float* g, const float* b,
                                             bf16_t* dst, int ldd, int lane, bool do_relu) {
  const int r = lane & 15, h = lane >> 4;
  const float* row = tile + r * DM + h * 64;
  float s1 = 0.f, s2 = 0.f;
#pragma unroll
  for (int j = 0; j < 64; j += 4) {
    float4 x = *(const float4*)(row + j);
    s1 += x.x + x.y + x.z + x.w;
    s2 += x.x * x.x + x.y * x.y + x.z * x.z + x.w * x.w;
  }
  s1 += __shfl_xor(s1, 16);
  s2 += __shfl_xor(s2, 16);
  const float mu   = s1 * (1.f / 128.f);
  const float rstd = rsqrtf(s2 * (1.f / 128.f) - mu * mu + GN_EPS);
  bf16_t* drow = dst + (long)r * ldd + h * 64;
  const float* gg = g + h * 64;
  const float* bb = b + h * 64;
#pragma unroll
  for (int j = 0; j < 64; j += 4) {
    float4 x  = *(const float4*)(row + j);
    float4 gv = *(const float4*)(gg + j);
    float4 bv = *(const float4*)(bb + j);
    bf16x4 o;
    float t;
    t = (x.x - mu) * rstd * gv.x + bv.x; if (do_relu) t = fmaxf(t, 0.f); o[0] = (bf16_t)t;
    t = (x.y - mu) * rstd * gv.y + bv.y; if (do_relu) t = fmaxf(t, 0.f); o[1] = (bf16_t)t;
    t = (x.z - mu) * rstd * gv.z + bv.z; if (do_relu) t = fmaxf(t, 0.f); o[2] = (bf16_t)t;
    t = (x.w - mu) * rstd * gv.w + bv.w; if (do_relu) t = fmaxf(t, 0.f); o[3] = (bf16_t)t;
    *(bf16x4*)(drow + j) = o;
  }
}

// ---------------------------------------------------------------------------
// fp32 -> bf16 bulk convert (n divisible by 4)
// ---------------------------------------------------------------------------
__global__ void cvt_f32_bf16(const float* __restrict__ s, bf16_t* __restrict__ d, int n4) {
  int i = blockIdx.x * blockDim.x + threadIdx.x;
  if (i < n4) {
    float4 v = ((const float4*)s)[i];
    union { bf16_t b[4]; unsigned long long u; } pk;
    pk.b[0] = (bf16_t)v.x; pk.b[1] = (bf16_t)v.y;
    pk.b[2] = (bf16_t)v.z; pk.b[3] = (bf16_t)v.w;
    ((unsigned long long*)d)[i] = pk.u;
  }
}

// ---------------------------------------------------------------------------
// Kernel A: q = bf16(relu(GN(actors @ Wq.T))), a0 = actors @ Wa.T (f32)
// 128 threads = 4 waves, 16 actor rows per wave.
// ---------------------------------------------------------------------------
__global__ void __launch_bounds__(128)
actor_pre_kernel(const bf16_t* __restrict__ actors_b,
                 const bf16_t* __restrict__ qW, const float* __restrict__ qg,
                 const float* __restrict__ qb,
                 const bf16_t* __restrict__ aW,
                 bf16_t* __restrict__ q_out, float* __restrict__ a0) {
  extern __shared__ char smem[];
  const int lane = threadIdx.x & 31, wave = threadIdx.x >> 5;
  float* tile = (float*)(smem + wave * 8192);
  const int rowBase = blockIdx.x * 64 + wave * 16;
  const bf16_t* abase = actors_b + (long)rowBase * DM;

  if (lane == 0) { __builtin_prefetch(qW, 0, 1); __builtin_prefetch(aW, 0, 1); }

  bf16x16 afr[4];
#pragma unroll
  for (int kc = 0; kc < 4; ++kc) afr[kc] = load_a_frag(abase, DM, kc * 32, lane);

  v8f acc[8]; const v8f z = {};
#pragma unroll
  for (int nt = 0; nt < 8; ++nt) acc[nt] = z;
#pragma unroll
  for (int kc = 0; kc < 4; ++kc)
#pragma unroll
    for (int nt = 0; nt < 8; ++nt)
      acc[nt] = wmma_bf16(afr[kc], load_b_frag(qW, DM, nt, kc * 32, lane), acc[nt]);
  stage_acc(tile, acc, lane);
  __syncthreads();
  gn_from_tile(tile, qg, qb, q_out + (long)rowBase * DM, DM, lane, true);
  __syncthreads();

#pragma unroll
  for (int nt = 0; nt < 8; ++nt) acc[nt] = z;
#pragma unroll
  for (int kc = 0; kc < 4; ++kc)
#pragma unroll
    for (int nt = 0; nt < 8; ++nt)
      acc[nt] = wmma_bf16(afr[kc], load_b_frag(aW, DM, nt, kc * 32, lane), acc[nt]);
  const int m0 = (lane >> 4) << 3, n0 = lane & 15;
#pragma unroll
  for (int nt = 0; nt < 8; ++nt)
#pragma unroll
    for (int v = 0; v < 8; ++v)
      a0[(long)(rowBase + m0 + v) * DM + (nt << 4) + n0] = acc[nt][v];
}

// ---------------------------------------------------------------------------
// Kernel B: full edge pipeline, 16 edges per wave, 4 waves per block.
// Random-row gathers (q[hi], nodes[wi]) are issued as async global->LDS copies
// up front and drained after GEMM1, overlapping gather latency with WMMA work.
// ---------------------------------------------------------------------------
__global__ void __launch_bounds__(128)
att_edge_kernel(const int* __restrict__ hi, const int* __restrict__ wi,
                const float* __restrict__ actor_ctrs, const float* __restrict__ node_ctrs,
                const float* __restrict__ d0W, const float* __restrict__ d0b,
                const bf16_t* __restrict__ d1W,
                const float* __restrict__ d1g, const float* __restrict__ d1b,
                const bf16_t* __restrict__ qrows, const bf16_t* __restrict__ nrows,
                const bf16_t* __restrict__ c0W,
                const float* __restrict__ c0g, const float* __restrict__ c0b,
                const bf16_t* __restrict__ c1W,
                float* __restrict__ a0) {
  extern __shared__ char smem[];
  const int lane = threadIdx.x & 31, wave = threadIdx.x >> 5;
  char* wsm = smem + wave * 20736;
  float*  tile = (float*)wsm;                       // 16x128 f32 staging tile
  bf16_t* mbuf = (bf16_t*)(wsm + 8192);             // 16x384 bf16: [d | q | c]
  int*    hiA  = (int*)(wsm + 8192 + 12288);
  int*    wiA  = hiA + 16;
  float*  dxy  = (float*)(wiA + 16);                // dx[16], dy[16]

  const long eBase = (long)blockIdx.x * 64 + wave * 16;
  if (lane < 16) {
    const int h = hi[eBase + lane], w = wi[eBase + lane];
    hiA[lane] = h; wiA[lane] = w;
    dxy[lane]      = actor_ctrs[2 * h]     - node_ctrs[2 * w];
    dxy[16 + lane] = actor_ctrs[2 * h + 1] - node_ctrs[2 * w + 1];
  }
  if (lane == 0) {
    __builtin_prefetch(d1W, 0, 1);
    __builtin_prefetch(c0W, 0, 1);
    __builtin_prefetch(c1W, 0, 1);
  }
  __syncthreads();

  // ---- issue async gathers: q[hi] -> mbuf[:,128:256], nodes[wi] -> [:,256:384].
  // These drain into LDS (ASYNCcnt) while GEMM1 runs below.
  {
    const unsigned mbase = (unsigned)((char*)mbuf - smem);  // dynamic LDS base = 0
#pragma unroll
    for (int it = 0; it < 8; ++it) {
      const int e   = it * 2 + (lane >> 4);
      const int col = (lane & 15) << 3;
      async_gather_b128(mbase + 2u * (unsigned)(e * K3 + 128 + col),
                        qrows + (long)hiA[e] * DM + col);
      async_gather_b128(mbase + 2u * (unsigned)(e * K3 + 256 + col),
                        nrows + (long)wiA[e] * DM + col);
    }
  }

  const int r  = lane & 15;
  const int hk = (lane >> 4) << 3;
  const float ddx = dxy[r], ddy = dxy[16 + r];

  v8f acc[8]; const v8f z = {};
#pragma unroll
  for (int nt = 0; nt < 8; ++nt) acc[nt] = z;

  // ---- stage 1 fused into GEMM1: hd = relu(d2 @ W0.T + b0) built directly in
  // A-fragment registers (K=2 rank update, VALU); then d_acc = hd @ W1.T.
#pragma unroll
  for (int kc = 0; kc < 4; ++kc) {
    bf16x16 a;
    const int kb = kc * 32 + hk;
#pragma unroll
    for (int j = 0; j < 8; ++j) {
      const int c0 = kb + j, c1 = kb + 16 + j;
      a[j]     = (bf16_t)fmaxf(ddx * d0W[2 * c0] + ddy * d0W[2 * c0 + 1] + d0b[c0], 0.f);
      a[j + 8] = (bf16_t)fmaxf(ddx * d0W[2 * c1] + ddy * d0W[2 * c1 + 1] + d0b[c1], 0.f);
    }
#pragma unroll
    for (int nt = 0; nt < 8; ++nt)
      acc[nt] = wmma_bf16(a, load_b_frag(d1W, DM, nt, kc * 32, lane), acc[nt]);
  }
  stage_acc(tile, acc, lane);
  __syncthreads();
  gn_from_tile(tile, d1g, d1b, mbuf, K3, lane, true);   // d -> mbuf[:,0:128]

  wait_async0();        // q/c rows now resident in LDS
  __syncthreads();

  // ---- GEMM2: m1 = relu(GN(m_in @ ctx0W.T)), K = 384
#pragma unroll
  for (int nt = 0; nt < 8; ++nt) acc[nt] = z;
#pragma unroll
  for (int kc = 0; kc < 12; ++kc) {
    const bf16x16 a = load_a_frag(mbuf, K3, kc * 32, lane);
#pragma unroll
    for (int nt = 0; nt < 8; ++nt)
      acc[nt] = wmma_bf16(a, load_b_frag(c0W, K3, nt, kc * 32, lane), acc[nt]);
  }
  stage_acc(tile, acc, lane);
  __syncthreads();
  gn_from_tile(tile, c0g, c0b, mbuf, K3, lane, true);   // m1 overwrites d region
  __syncthreads();

  // ---- GEMM3: m2 = m1 @ ctx1W.T, then scatter-add into a0[hi]
#pragma unroll
  for (int nt = 0; nt < 8; ++nt) acc[nt] = z;
#pragma unroll
  for (int kc = 0; kc < 4; ++kc) {
    const bf16x16 a = load_a_frag(mbuf, K3, kc * 32, lane);
#pragma unroll
    for (int nt = 0; nt < 8; ++nt)
      acc[nt] = wmma_bf16(a, load_b_frag(c1W, DM, nt, kc * 32, lane), acc[nt]);
  }
  const int m0 = (lane >> 4) << 3, n0 = lane & 15;
#pragma unroll
  for (int nt = 0; nt < 8; ++nt) {
    const int n = (nt << 4) + n0;
#pragma unroll
    for (int v = 0; v < 8; ++v) {
      const long rowi = hiA[m0 + v];
      __hip_atomic_fetch_add(&a0[rowi * DM + n], acc[nt][v],
                             __ATOMIC_RELAXED, __HIP_MEMORY_SCOPE_AGENT);
    }
  }
}

// ---------------------------------------------------------------------------
// Kernel C: a = relu(GN(a0)); a = GN(a @ Wl.T); out = relu(a + res)
// ---------------------------------------------------------------------------
__global__ void __launch_bounds__(128)
actor_post_kernel(const float* __restrict__ a0,
                  const float* __restrict__ ng, const float* __restrict__ nb,
                  const bf16_t* __restrict__ lW,
                  const float* __restrict__ lg, const float* __restrict__ lb,
                  const float* __restrict__ res,
                  float* __restrict__ out_f, bf16_t* __restrict__ out_b) {
  extern __shared__ char smem[];
  const int lane = threadIdx.x & 31, wave = threadIdx.x >> 5;
  char* wsm = smem + wave * 12288;
  bf16_t* abuf = (bf16_t*)wsm;               // 16x128 bf16 GEMM input
  float*  tile = (float*)(wsm + 4096);       // 16x128 f32 staging
  const int rowBase = blockIdx.x * 64 + wave * 16;
  const int r = lane & 15, h = lane >> 4;

  // relu(GN(a0)) -> abuf (bf16)
  const float* grow = a0 + (long)(rowBase + r) * DM + h * 64;
  float s1 = 0.f, s2 = 0.f;
#pragma unroll
  for (int j = 0; j < 64; j += 4) {
    float4 x = *(const float4*)(grow + j);
    s1 += x.x + x.y + x.z + x.w;
    s2 += x.x * x.x + x.y * x.y + x.z * x.z + x.w * x.w;
  }
  s1 += __shfl_xor(s1, 16); s2 += __shfl_xor(s2, 16);
  float mu   = s1 * (1.f / 128.f);
  float rstd = rsqrtf(s2 * (1.f / 128.f) - mu * mu + GN_EPS);
  {
    bf16_t* drow = abuf + r * DM + h * 64;
    const float* gg = ng + h * 64; const float* bb = nb + h * 64;
#pragma unroll
    for (int j = 0; j < 64; j += 4) {
      float4 x  = *(const float4*)(grow + j);
      float4 gv = *(const float4*)(gg + j);
      float4 bv = *(const float4*)(bb + j);
      bf16x4 o;
      o[0] = (bf16_t)fmaxf((x.x - mu) * rstd * gv.x + bv.x, 0.f);
      o[1] = (bf16_t)fmaxf((x.y - mu) * rstd * gv.y + bv.y, 0.f);
      o[2] = (bf16_t)fmaxf((x.z - mu) * rstd * gv.z + bv.z, 0.f);
      o[3] = (bf16_t)fmaxf((x.w - mu) * rstd * gv.w + bv.w, 0.f);
      *(bf16x4*)(drow + j) = o;
    }
  }
  __syncthreads();

  // GEMM lin
  v8f acc[8]; const v8f z = {};
#pragma unroll
  for (int nt = 0; nt < 8; ++nt) acc[nt] = z;
#pragma unroll
  for (int kc = 0; kc < 4; ++kc) {
    const bf16x16 a = load_a_frag(abuf, DM, kc * 32, lane);
#pragma unroll
    for (int nt = 0; nt < 8; ++nt)
      acc[nt] = wmma_bf16(a, load_b_frag(lW, DM, nt, kc * 32, lane), acc[nt]);
  }
  stage_acc(tile, acc, lane);
  __syncthreads();

  // GN (no relu) + residual + relu -> outputs
  const float* row = tile + r * DM + h * 64;
  s1 = 0.f; s2 = 0.f;
#pragma unroll
  for (int j = 0; j < 64; j += 4) {
    float4 x = *(const float4*)(row + j);
    s1 += x.x + x.y + x.z + x.w;
    s2 += x.x * x.x + x.y * x.y + x.z * x.z + x.w * x.w;
  }
  s1 += __shfl_xor(s1, 16); s2 += __shfl_xor(s2, 16);
  mu   = s1 * (1.f / 128.f);
  rstd = rsqrtf(s2 * (1.f / 128.f) - mu * mu + GN_EPS);
  const float* rrow = res + (long)(rowBase + r) * DM + h * 64;
  float* orow = out_f + (long)(rowBase + r) * DM + h * 64;
  const float* gg = lg + h * 64; const float* bb = lb + h * 64;
#pragma unroll
  for (int j = 0; j < 64; j += 4) {
    float4 x  = *(const float4*)(row + j);
    float4 gv = *(const float4*)(gg + j);
    float4 bv = *(const float4*)(bb + j);
    float4 rv = *(const float4*)(rrow + j);
    float4 o;
    o.x = fmaxf((x.x - mu) * rstd * gv.x + bv.x + rv.x, 0.f);
    o.y = fmaxf((x.y - mu) * rstd * gv.y + bv.y + rv.y, 0.f);
    o.z = fmaxf((x.z - mu) * rstd * gv.z + bv.z + rv.z, 0.f);
    o.w = fmaxf((x.w - mu) * rstd * gv.w + bv.w + rv.w, 0.f);
    *(float4*)(orow + j) = o;
    if (out_b) {
      bf16x4 ob;
      ob[0] = (bf16_t)o.x; ob[1] = (bf16_t)o.y; ob[2] = (bf16_t)o.z; ob[3] = (bf16_t)o.w;
      *(bf16x4*)(out_b + (long)(rowBase + r) * DM + h * 64 + j) = ob;
    }
  }
}

// ---------------------------------------------------------------------------
// Host launcher
// ---------------------------------------------------------------------------
extern "C" void kernel_launch(void* const* d_in, const int* in_sizes, int n_in,
                              void* d_out, int out_size, void* d_ws, size_t ws_size,
                              hipStream_t stream) {
  (void)in_sizes; (void)n_in; (void)out_size;
  const int Na = 8192, Nn = 65536, E = 262144;

  const float* actors     = (const float*)d_in[0];
  const float* nodes      = (const float*)d_in[1];
  const float* actor_ctrs = (const float*)d_in[2];
  const float* node_ctrs  = (const float*)d_in[3];
  const int*   hi         = (const int*)d_in[4];
  const int*   wi         = (const int*)d_in[5];
  const float* dist0_W    = (const float*)d_in[6];
  const float* dist0_b    = (const float*)d_in[7];
  const float* dist1_W    = (const float*)d_in[8];
  const float* dist1_g    = (const float*)d_in[9];
  const float* dist1_b    = (const float*)d_in[10];
  const float* query_W    = (const float*)d_in[11];
  const float* query_g    = (const float*)d_in[12];
  const float* query_b    = (const float*)d_in[13];
  const float* ctx0_W     = (const float*)d_in[14];
  const float* ctx0_g     = (const float*)d_in[15];
  const float* ctx0_b     = (const float*)d_in[16];
  const float* ctx1_W     = (const float*)d_in[17];
  const float* agt_W      = (const float*)d_in[18];
  const float* norm_g     = (const float*)d_in[19];
  const float* norm_b     = (const float*)d_in[20];
  const float* lin_W      = (const float*)d_in[21];
  const float* lin_g      = (const float*)d_in[22];
  const float* lin_b      = (const float*)d_in[23];
  float* outp = (float*)d_out;

  // workspace carve (256B aligned)
  char* w = (char*)d_ws;
  auto take = [&](size_t bytes) -> char* {
    char* p = w; w += (bytes + 255) & ~(size_t)255; return p;
  };
  bf16_t* nodes_bb  = (bf16_t*)take((size_t)Nn * DM * 2);
  bf16_t* actors_bb = (bf16_t*)take((size_t)Na * DM * 2);
  bf16_t* q_bb      = (bf16_t*)take((size_t)Na * DM * 2);
  float*  a0        = (float*)take((size_t)Na * DM * 4);
  float*  act_tmp   = (float*)take((size_t)Na * DM * 4);
  bf16_t* dist1_Wb  = (bf16_t*)take((size_t)2 * DM * DM * 2);
  bf16_t* query_Wb  = (bf16_t*)take((size_t)2 * DM * DM * 2);
  bf16_t* ctx0_Wb   = (bf16_t*)take((size_t)2 * DM * K3 * 2);
  bf16_t* ctx1_Wb   = (bf16_t*)take((size_t)2 * DM * DM * 2);
  bf16_t* agt_Wb    = (bf16_t*)take((size_t)2 * DM * DM * 2);
  bf16_t* lin_Wb    = (bf16_t*)take((size_t)2 * DM * DM * 2);
  if ((size_t)(w - (char*)d_ws) > ws_size) return;  // insufficient scratch

  auto cvt = [&](const float* s, bf16_t* d, long n) {
    int n4 = (int)(n / 4);
    cvt_f32_bf16<<<(n4 + 255) / 256, 256, 0, stream>>>(s, d, n4);
  };
  cvt(nodes,   nodes_bb,  (long)Nn * DM);
  cvt(actors,  actors_bb, (long)Na * DM);
  cvt(dist1_W, dist1_Wb,  2L * DM * DM);
  cvt(query_W, query_Wb,  2L * DM * DM);
  cvt(ctx0_W,  ctx0_Wb,   2L * DM * K3);
  cvt(ctx1_W,  ctx1_Wb,   2L * DM * DM);
  cvt(agt_W,   agt_Wb,    2L * DM * DM);
  cvt(lin_W,   lin_Wb,    2L * DM * DM);

  const size_t PRE_SMEM  = 4 * 8192;
  const size_t EDGE_SMEM = 4 * 20736;
  const size_t POST_SMEM = 4 * 12288;

  for (int i = 0; i < 2; ++i) {
    const long oDD = (long)i * DM * DM, oD3 = (long)i * DM * K3, oD = (long)i * DM;

    actor_pre_kernel<<<Na / 64, 128, PRE_SMEM, stream>>>(
        actors_bb, query_Wb + oDD, query_g + oD, query_b + oD,
        agt_Wb + oDD, q_bb, a0);

    att_edge_kernel<<<E / 64, 128, EDGE_SMEM, stream>>>(
        hi, wi, actor_ctrs, node_ctrs,
        dist0_W + (long)i * DM * 2, dist0_b + oD,
        dist1_Wb + oDD, dist1_g + oD, dist1_b + oD,
        q_bb, nodes_bb,
        ctx0_Wb + oD3, ctx0_g + oD, ctx0_b + oD,
        ctx1_Wb + oDD, a0);

    const float* res  = (i == 0) ? actors : act_tmp;
    float*       outf = (i == 1) ? outp : act_tmp;
    bf16_t*      outb = (i == 1) ? (bf16_t*)nullptr : actors_bb;
    actor_post_kernel<<<Na / 64, 128, POST_SMEM, stream>>>(
        a0, norm_g + oD, norm_b + oD,
        lin_Wb + oDD, lin_g + oD, lin_b + oD,
        res, outf, outb);
  }
}
